// MultiScaleBlock_69595650065158
// MI455X (gfx1250) — matrix operand order repaired and checked by reading, not compiled, not run on hardware
//
#include <hip/hip_runtime.h>
#include <hip/hip_bf16.h>
#include <math.h>

typedef __attribute__((ext_vector_type(16))) __bf16 v16bf;
typedef __attribute__((ext_vector_type(8)))  float  v8f;

union Frag {
    v16bf    v;
    uint32_t u[8];
    uint4    q[2];
    __bf16   h[16];
};

static __device__ __forceinline__ float  bf2f(__bf16 h) { return (float)h; }
static __device__ __forceinline__ __bf16 f2bf(float f)  { return (__bf16)f; }

// ---------------------------------------------------------------------------
// Weight pre-pack: f32 W[K,N] -> bf16 fragments in this file's WMMA B-operand k order (contiguous, k = 16*half + i).
// Fragment (tn, kb) holds the 32x16 panel B[kb*32 .. +31][tn*16 .. +15].
// Lane (half*16+l16), element i  <->  B[kb*32 + half*16 + i][tn*16 + l16].
// Flat: out[((tn*(K/32)+kb)*32 + lane)*16 + i]
// ---------------------------------------------------------------------------
__global__ void pack_b_k(const float* __restrict__ W, __bf16* __restrict__ out,
                         int K, int N) {
    int o = blockIdx.x * blockDim.x + threadIdx.x;
    if (o >= K * N) return;
    int i    = o & 15;
    int lane = (o >> 4) & 31;
    int idx  = o >> 9;
    int kblocks = K >> 5;
    int kb = idx % kblocks;
    int tn = idx / kblocks;
    int k = kb * 32 + (lane >> 4) * 16 + i;
    int n = tn * 16 + (lane & 15);
    out[o] = f2bf(W[(size_t)k * N + n]);
}

// window-partition row map: window-order token g -> spatial row in 56x56 grid
__global__ void build_amap_k(int* __restrict__ amap) {
    int g = blockIdx.x * blockDim.x + threadIdx.x;
    if (g >= 100352) return;
    int bn = g >> 6, n = g & 63;
    int b = bn / 49, w = bn % 49, wi = w / 7, wj = w % 7;
    int r = n >> 3, c = n & 7;
    amap[g] = (b * 56 + wi * 8 + r) * 56 + wj * 8 + c;
}

// window-unpartition row map: window-order pooled token t -> spatial row (28x28)
__global__ void build_cmap_k(int* __restrict__ cmap) {
    int t = blockIdx.x * blockDim.x + threadIdx.x;
    if (t >= 25088) return;
    int bn = t >> 4, nq = t & 15;
    int b = bn / 49, w = bn % 49, wi = w / 7, wj = w % 7;
    int r = nq >> 2, c = nq & 3;
    cmap[t] = (b * 28 + wi * 4 + r) * 28 + wj * 4 + c;
}

// ---------------------------------------------------------------------------
// LayerNorm: one wave per row, wave32 shuffle reduction, bf16 output
// ---------------------------------------------------------------------------
template <int C>
__global__ void layernorm_bf16_k(const float* __restrict__ x,
                                 const float* __restrict__ g,
                                 const float* __restrict__ b,
                                 __bf16* __restrict__ out, int rows) {
    int wave = (blockIdx.x * blockDim.x + threadIdx.x) >> 5;
    if (wave >= rows) return;
    int lane = threadIdx.x & 31;
    const float* xr = x + (size_t)wave * C;
    float s = 0.f, sq = 0.f;
    #pragma unroll
    for (int i = lane; i < C; i += 32) { float v = xr[i]; s += v; sq += v * v; }
    #pragma unroll
    for (int o = 16; o > 0; o >>= 1) { s += __shfl_xor(s, o); sq += __shfl_xor(sq, o); }
    float mean = s / (float)C;
    float var  = sq / (float)C - mean * mean;
    float inv  = rsqrtf(var + 1e-6f);
    __bf16* orow = out + (size_t)wave * C;
    #pragma unroll
    for (int i = lane; i < C; i += 32)
        orow[i] = f2bf((xr[i] - mean) * inv * g[i] + b[i]);
}

// ---------------------------------------------------------------------------
// bf16 WMMA GEMM, one wave computes a 16x64 strip (4 N-tiles share the A frag).
// B comes pre-packed in WMMA fragment order -> two b128 loads per fragment.
// Epilogues: 0 bias->f32, 1 bias->bf16, 2 bias+res (row-mapped)->f32,
//            3 bias+GELU->bf16, 4 bias+res->f32
// ---------------------------------------------------------------------------
template <int EPI>
__global__ void gemm_bf16_wmma_k(const __bf16* __restrict__ A,
                                 const __bf16* __restrict__ Bp,   // packed
                                 const float* __restrict__ bias,
                                 const int* __restrict__ amap,
                                 const int* __restrict__ cmap,
                                 const float* __restrict__ res,
                                 float* __restrict__ Cf,
                                 __bf16* __restrict__ Cb,
                                 int M, int N, int K) {
    const int wid = (blockIdx.x * blockDim.x + threadIdx.x) >> 5;
    const int ntn = N >> 6;                    // 64-wide column groups
    const int ntiles = (M >> 4) * ntn;
    if (wid >= ntiles) return;
    const int tm = wid / ntn, tn = wid % ntn;
    const int m0 = tm << 4, n0 = tn << 6;
    const int lane = threadIdx.x & 31;
    const int l16 = lane & 15, half = lane >> 4;
    const int kblocks = K >> 5;

    const int arow = m0 + l16;
    const int asrc = amap ? amap[arow] : arow;
    const uint32_t* Arow = (const uint32_t*)(A + (size_t)asrc * K);

    v8f zero = {};
    v8f acc[4];
    #pragma unroll
    for (int t = 0; t < 4; ++t) acc[t] = zero;

    for (int kb = 0; kb < kblocks; ++kb) {
        int k0 = kb << 5;
        Frag a;
        // A 16x32 fragment: lane row m0+l16; K run half*16+0..15
        int base = (k0 + half * 16) >> 1;
        a.q[0] = *(const uint4*)(Arow + base);
        base = (k0 + half * 16 + 8) >> 1;
        a.q[1] = *(const uint4*)(Arow + base);
        #pragma unroll
        for (int t = 0; t < 4; ++t) {
            Frag bf;
            const uint4* bp = (const uint4*)(Bp +
                ((size_t)(((tn << 2) + t) * kblocks + kb) * 32 + lane) * 16);
            bf.q[0] = bp[0];
            bf.q[1] = bp[1];
            acc[t] = __builtin_amdgcn_wmma_f32_16x16x32_bf16(false, a.v, false, bf.v,
                                                             (short)0, acc[t], false, false);
        }
    }

    #pragma unroll
    for (int t = 0; t < 4; ++t) {
        const int col = n0 + (t << 4) + l16;
        const float bv = bias[col];
        #pragma unroll
        for (int r = 0; r < 8; ++r) {
            int row = m0 + r + 8 * half;
            float vacc = acc[t][r] + bv;
            if (EPI == 0) {
                Cf[(size_t)row * N + col] = vacc;
            } else if (EPI == 1) {
                Cb[(size_t)row * N + col] = f2bf(vacc);
            } else if (EPI == 2) {
                int orow = cmap[row];
                size_t o = (size_t)orow * N + col;
                Cf[o] = res[o] + vacc;
            } else if (EPI == 3) {
                float gl = 0.5f * vacc * (1.0f + erff(vacc * 0.70710678118654752f));
                Cb[(size_t)row * N + col] = f2bf(gl);
            } else {
                size_t o = (size_t)row * N + col;
                Cf[o] = res[o] + vacc;
            }
        }
    }
}

// ---------------------------------------------------------------------------
// 2x2 maxpool of proj output (32,56,56,384)->(32,28,28,384), f32
// ---------------------------------------------------------------------------
__global__ void maxpool_proj_k(const float* __restrict__ in, float* __restrict__ out) {
    int e = blockIdx.x * blockDim.x + threadIdx.x;
    const int total = 32 * 28 * 28 * 384;
    if (e >= total) return;
    int c = e % 384, t = e / 384;
    int j = t % 28; t /= 28;
    int i = t % 28; int b = t / 28;
    int r0 = (b * 56 + 2 * i) * 56 + 2 * j;
    int r1 = r0 + 56;
    float v = fmaxf(fmaxf(in[(size_t)r0 * 384 + c], in[(size_t)(r0 + 1) * 384 + c]),
                    fmaxf(in[(size_t)r1 * 384 + c], in[(size_t)(r1 + 1) * 384 + c]));
    out[e] = v;
}

// 2x2 maxpool of q slice of qkv (window order) -> q_bf16 [1568*16, 384]
__global__ void maxpool_q_k(const __bf16* __restrict__ qkv, __bf16* __restrict__ q) {
    int e = blockIdx.x * blockDim.x + threadIdx.x;
    const int total = 1568 * 16 * 384;
    if (e >= total) return;
    int c = e % 384, t = e / 384;
    int nq = t & 15, bn = t >> 4;
    int ri = nq >> 2, ci = nq & 3;
    size_t base = (size_t)bn * 64;
    float m = -1e30f;
    #pragma unroll
    for (int dr = 0; dr < 2; ++dr)
        #pragma unroll
        for (int dc = 0; dc < 2; ++dc) {
            int n = (2 * ri + dr) * 8 + (2 * ci + dc);
            m = fmaxf(m, bf2f(qkv[(base + n) * 1152 + c]));
        }
    q[(size_t)t * 384 + c] = f2bf(m);
}

// ---------------------------------------------------------------------------
// Fused attention: one wave per (window, head). q(16x96)*k^T(96x64) -> softmax
// -> LDS round trip (C-layout -> A-layout) -> attn(16x64)*v(64x96).
// ---------------------------------------------------------------------------
__global__ void attention_k(const __bf16* __restrict__ q,    // [25088, 384]
                            const __bf16* __restrict__ qkv,  // [100352, 1152]
                            __bf16* __restrict__ out) {      // [25088, 384]
    __shared__ __bf16 smem[8][16 * 64];
    int task = (blockIdx.x * blockDim.x + threadIdx.x) >> 5;
    const int ntasks = 1568 * 4;
    if (task >= ntasks) task = ntasks - 1;   // keep all waves alive for __syncthreads
    int h = task & 3, bn = task >> 2;
    int lane = threadIdx.x & 31, l16 = lane & 15, half = lane >> 4;
    int w = threadIdx.x >> 5;
    const float scale = 0.102062072615966f;  // 96^-0.5

    // ---- phase 1: attn = q @ k^T  (M=16 queries, N=64 keys, K=96 dims) ----
    v8f zero = {};
    v8f acc[4];
    #pragma unroll
    for (int t = 0; t < 4; ++t) acc[t] = zero;

    const uint32_t* Qrow = (const uint32_t*)(q + ((size_t)bn * 16 + l16) * 384 + h * 96);
    for (int k0 = 0; k0 < 96; k0 += 32) {
        Frag a;
        int base = (k0 + half * 16) >> 1;
        a.q[0] = *(const uint4*)(Qrow + base);
        base = (k0 + half * 16 + 8) >> 1;
        a.q[1] = *(const uint4*)(Qrow + base);
        #pragma unroll
        for (int t = 0; t < 4; ++t) {
            // B col = key token (t*16+l16); K run = head dims k0+half*16+0..15
            Frag bf;
            const uint4* Krow = (const uint4*)(qkv +
                ((size_t)bn * 64 + t * 16 + l16) * 1152 + 384 + h * 96 + k0 + half * 16);
            bf.q[0] = Krow[0];
            bf.q[1] = Krow[1];
            acc[t] = __builtin_amdgcn_wmma_f32_16x16x32_bf16(false, a.v, false, bf.v,
                                                             (short)0, acc[t], false, false);
        }
    }

    // ---- softmax over 64 keys per query row (rows live in 16-lane halves) ----
    float pr[4][8];
    #pragma unroll
    for (int r = 0; r < 8; ++r) {
        float mx = -1e30f;
        #pragma unroll
        for (int t = 0; t < 4; ++t) mx = fmaxf(mx, acc[t][r] * scale);
        #pragma unroll
        for (int o = 1; o < 16; o <<= 1) mx = fmaxf(mx, __shfl_xor(mx, o));
        float s = 0.f;
        #pragma unroll
        for (int t = 0; t < 4; ++t) {
            float e = __expf(acc[t][r] * scale - mx);
            pr[t][r] = e; s += e;
        }
        #pragma unroll
        for (int o = 1; o < 16; o <<= 1) s += __shfl_xor(s, o);
        float inv = 1.f / s;
        #pragma unroll
        for (int t = 0; t < 4; ++t) pr[t][r] *= inv;
    }

    // C-layout -> LDS (16x64 bf16 tile)
    #pragma unroll
    for (int r = 0; r < 8; ++r)
        #pragma unroll
        for (int t = 0; t < 4; ++t)
            smem[w][(r + 8 * half) * 64 + t * 16 + l16] = f2bf(pr[t][r]);
    __syncthreads();

    // ---- phase 2: out = attn @ v  (M=16, N=96, K=64) ----
    v8f acc2[6];
    #pragma unroll
    for (int t = 0; t < 6; ++t) acc2[t] = zero;
    const uint32_t* Srow = (const uint32_t*)&smem[w][l16 * 64];
    for (int k0 = 0; k0 < 64; k0 += 32) {
        Frag a;
        int base = (k0 + half * 16) >> 1;
        #pragma unroll
        for (int j = 0; j < 4; ++j) a.u[j] = Srow[base + j];
        base = (k0 + half * 16 + 8) >> 1;
        #pragma unroll
        for (int j = 0; j < 4; ++j) a.u[4 + j] = Srow[base + j];
        #pragma unroll
        for (int t = 0; t < 6; ++t) {
            Frag bf;
            #pragma unroll
            for (int j = 0; j < 8; ++j) {
                int kk = k0 + half * 16 + 2 * j;
                const __bf16* vp = qkv + ((size_t)bn * 64 + kk) * 1152 + 768 + h * 96 + t * 16 + l16;
                union { uint32_t u; __bf16 hh[2]; } p;
                p.hh[0] = vp[0];
                p.hh[1] = vp[1152];
                bf.u[j] = p.u;
            }
            acc2[t] = __builtin_amdgcn_wmma_f32_16x16x32_bf16(false, a.v, false, bf.v,
                                                              (short)0, acc2[t], false, false);
        }
    }
    #pragma unroll
    for (int t = 0; t < 6; ++t)
        #pragma unroll
        for (int r = 0; r < 8; ++r)
            out[((size_t)bn * 16 + r + 8 * half) * 384 + h * 96 + t * 16 + l16] = f2bf(acc2[t][r]);
}

// ---------------------------------------------------------------------------
// host side
// ---------------------------------------------------------------------------
static inline int gemm_blocks(int M, int N) {
    long long tiles = (long long)(M >> 4) * (N >> 6);
    return (int)((tiles * 32 + 255) / 256);
}

extern "C" void kernel_launch(void* const* d_in, const int* in_sizes, int n_in,
                              void* d_out, int out_size, void* d_ws, size_t ws_size,
                              hipStream_t stream) {
    const float* x      = (const float*)d_in[0];
    const float* n1g    = (const float*)d_in[1];
    const float* n1b    = (const float*)d_in[2];
    const float* proj_w = (const float*)d_in[3];
    const float* proj_b = (const float*)d_in[4];
    const float* qkv_w  = (const float*)d_in[5];
    const float* qkv_b  = (const float*)d_in[6];
    const float* apw    = (const float*)d_in[7];
    const float* apb    = (const float*)d_in[8];
    const float* n2g    = (const float*)d_in[9];
    const float* n2b    = (const float*)d_in[10];
    const float* fc1w   = (const float*)d_in[11];
    const float* fc1b   = (const float*)d_in[12];
    const float* fc2w   = (const float*)d_in[13];
    const float* fc2b   = (const float*)d_in[14];
    float* out = (float*)d_out;

    char* p = (char*)d_ws;
    auto alloc = [&](size_t bytes) -> char* {
        char* r = p;
        p += (bytes + 255) & ~(size_t)255;
        return r;
    };
    __bf16* wproj = (__bf16*)alloc((size_t)192 * 384 * 2);
    __bf16* wqkv  = (__bf16*)alloc((size_t)192 * 1152 * 2);
    __bf16* wap   = (__bf16*)alloc((size_t)384 * 384 * 2);
    __bf16* wfc1  = (__bf16*)alloc((size_t)384 * 1536 * 2);
    __bf16* wfc2  = (__bf16*)alloc((size_t)1536 * 384 * 2);
    int*    amap  = (int*)alloc((size_t)100352 * 4);
    int*    cmap  = (int*)alloc((size_t)25088 * 4);
    __bf16* xn    = (__bf16*)alloc((size_t)100352 * 192 * 2);
    float*  ptmp  = (float*)alloc((size_t)100352 * 384 * 4);
    float*  shortcut = (float*)alloc((size_t)25088 * 384 * 4);
    __bf16* qkvb  = (__bf16*)alloc((size_t)100352 * 1152 * 2);
    __bf16* qb    = (__bf16*)alloc((size_t)25088 * 384 * 2);
    __bf16* attno = (__bf16*)alloc((size_t)25088 * 384 * 2);
    float*  x2    = (float*)alloc((size_t)25088 * 384 * 4);
    __bf16* x2n   = (__bf16*)alloc((size_t)25088 * 384 * 2);
    __bf16* h1    = (__bf16*)alloc((size_t)25088 * 1536 * 2);

    // weight packing (WMMA B-fragment order) + index maps
    auto pack = [&](const float* s, __bf16* d, int K, int N) {
        int n = K * N;
        pack_b_k<<<(n + 255) / 256, 256, 0, stream>>>(s, d, K, N);
    };
    pack(proj_w, wproj, 192, 384);
    pack(qkv_w,  wqkv,  192, 1152);
    pack(apw,    wap,   384, 384);
    pack(fc1w,   wfc1,  384, 1536);
    pack(fc2w,   wfc2,  1536, 384);
    build_amap_k<<<(100352 + 255) / 256, 256, 0, stream>>>(amap);
    build_cmap_k<<<(25088 + 255) / 256, 256, 0, stream>>>(cmap);

    // LN1 (100352 rows of 192) -> bf16
    layernorm_bf16_k<192><<<100352 / 8, 256, 0, stream>>>(x, n1g, n1b, xn, 100352);

    // proj GEMM (f32 out) then 2x2 maxpool -> shortcut
    gemm_bf16_wmma_k<0><<<gemm_blocks(100352, 384), 256, 0, stream>>>(
        xn, wproj, proj_b, nullptr, nullptr, nullptr, ptmp, nullptr, 100352, 384, 192);
    maxpool_proj_k<<<(32 * 28 * 28 * 384 + 255) / 256, 256, 0, stream>>>(ptmp, shortcut);

    // qkv GEMM (window-gathered A rows, bf16 out in window order)
    gemm_bf16_wmma_k<1><<<gemm_blocks(100352, 1152), 256, 0, stream>>>(
        xn, wqkv, qkv_b, amap, nullptr, nullptr, nullptr, qkvb, 100352, 1152, 192);

    // pooled q
    maxpool_q_k<<<(1568 * 16 * 384 + 255) / 256, 256, 0, stream>>>(qkvb, qb);

    // attention: 1568 windows * 4 heads, 8 waves/block
    attention_k<<<(1568 * 4) / 8, 256, 0, stream>>>(qb, qkvb, attno);

    // attn output proj + window-unpartition + shortcut add -> x2
    gemm_bf16_wmma_k<2><<<gemm_blocks(25088, 384), 256, 0, stream>>>(
        attno, wap, apb, nullptr, cmap, shortcut, x2, nullptr, 25088, 384, 384);

    // LN2 -> bf16
    layernorm_bf16_k<384><<<25088 / 8, 256, 0, stream>>>(x2, n2g, n2b, x2n, 25088);

    // MLP: fc1 + GELU -> bf16, fc2 + residual -> d_out
    gemm_bf16_wmma_k<3><<<gemm_blocks(25088, 1536), 256, 0, stream>>>(
        x2n, wfc1, fc1b, nullptr, nullptr, nullptr, nullptr, h1, 25088, 1536, 384);
    gemm_bf16_wmma_k<4><<<gemm_blocks(25088, 384), 256, 0, stream>>>(
        h1, wfc2, fc2b, nullptr, nullptr, x2, out, nullptr, 25088, 384, 1536);
}
